// Parsimony_70935679861196
// MI455X (gfx1250) — 8-bit matrix operand order repaired and checked by reading, compiled, not run on hardware
//
#include <hip/hip_runtime.h>
#include <stdint.h>

#define NTIPS   128
#define NSITES  1024
#define BS      32
#define NSTEPS  (NTIPS - 1)
#define NNODES  (2 * NTIPS - 1)
#define NWORDS  (NSITES / 32)
#define ROWU32  (5 * NWORDS)
#define TIPROWS (NTIPS * 5)

typedef __attribute__((ext_vector_type(8))) int v8i;
typedef int vec4i __attribute__((vector_size(16)));

#define AS_GLOBAL __attribute__((address_space(1)))
#define AS_LOCAL  __attribute__((address_space(3)))

__device__ __forceinline__ void async_copy_b128(const void* gsrc, void* ldst) {
#if __has_builtin(__builtin_amdgcn_global_load_async_to_lds_b128)
    __builtin_amdgcn_global_load_async_to_lds_b128(
        (AS_GLOBAL vec4i*)gsrc, (AS_LOCAL vec4i*)ldst,  0,  0);
#else
    asm volatile("global_load_async_to_lds_b128 %0, %1, off"
                 :: "v"((uint32_t)(uintptr_t)ldst), "v"(gsrc)
                 : "memory");
#endif
}

__device__ __forceinline__ void wait_asynccnt0() {
#if __has_builtin(__builtin_amdgcn_s_wait_asynccnt)
    __builtin_amdgcn_s_wait_asynccnt(0);
#else
    asm volatile("s_wait_asynccnt 0" ::: "memory");
#endif
}

__global__ void pack_tips(const int* __restrict__ L, uint32_t* __restrict__ packedL) {
    int gid = blockIdx.x * blockDim.x + threadIdx.x;
    if (gid >= TIPROWS * NWORDS) return;
    const int* src = L + gid * 32;
    uint32_t m = 0;
#pragma unroll
    for (int j = 0; j < 32; ++j) m |= (src[j] != 0 ? 1u : 0u) << j;
    packedL[gid] = m;
}

__global__ void fitch_steps(const uint32_t* __restrict__ packedL,
                            const int* __restrict__ pairs,
                            uint8_t* __restrict__ scoreB) {
    extern __shared__ uint32_t st[];
    const int b = blockIdx.x;
    const int t = threadIdx.x;

    for (int r = 0; r < (TIPROWS * NWORDS) / (4 * 32); ++r) {
        const int w = (r * 32 + t) * 4;
        async_copy_b128(packedL + w, st + w);
    }
    wait_asynccnt0();

    uint32_t cnt[8] = {0, 0, 0, 0, 0, 0, 0, 0};
    const int* pb = pairs + b * 3;

    for (int i = 0; i < NSTEPS; ++i) {
        const int* pr = pb + i * (BS * 3);
        __builtin_prefetch(pr + BS * 3, 0, 1);
        const int c0 = pr[0], c1 = pr[1], pp = pr[2];

        const uint32_t* A = st + c0 * ROWU32 + t;
        const uint32_t* B = st + c1 * ROWU32 + t;
        uint32_t*       P = st + pp * ROWU32 + t;

        const uint32_t a0 = A[0 * NWORDS], a1 = A[1 * NWORDS], a2 = A[2 * NWORDS],
                       a3 = A[3 * NWORDS], a4 = A[4 * NWORDS];
        const uint32_t b0 = B[0 * NWORDS], b1 = B[1 * NWORDS], b2 = B[2 * NWORDS],
                       b3 = B[3 * NWORDS], b4 = B[4 * NWORDS];

        const uint32_t n0 = a0 & b0, n1 = a1 & b1, n2 = a2 & b2,
                       n3 = a3 & b3, n4 = a4 & b4;
        const uint32_t em = ~(n0 | n1 | n2 | n3 | n4);

        P[0 * NWORDS] = n0 | (em & (a0 | b0));
        P[1 * NWORDS] = n1 | (em & (a1 | b1));
        P[2 * NWORDS] = n2 | (em & (a2 | b2));
        P[3 * NWORDS] = n3 | (em & (a3 | b3));
        P[4 * NWORDS] = n4 | (em & (a4 | b4));

#pragma unroll
        for (int w = 0; w < 8; ++w) {
            const uint32_t nib = (em >> (4 * w)) & 0xFu;
            cnt[w] += (nib & 1u) | ((nib & 2u) << 7) | ((nib & 4u) << 14) |
                      ((nib & 8u) << 21);
        }
    }

    uint8_t* row = scoreB + b * NSITES + t * 32;
    ((uint4*)row)[0] = make_uint4(cnt[0], cnt[1], cnt[2], cnt[3]);
    ((uint4*)row)[1] = make_uint4(cnt[4], cnt[5], cnt[6], cnt[7]);
}

__global__ void wmma_reduce(const uint8_t* __restrict__ scoreB,
                            const int* __restrict__ counts,
                            float* __restrict__ out) {
    const int lane = threadIdx.x & 31;
    const int tile = threadIdx.x >> 5;
    const int half = (lane >> 4) & 1;
    const int m    = tile * 16 + (lane & 15);

    v8i acc = {0, 0, 0, 0, 0, 0, 0, 0};

    for (int kc = 0; kc < 16; ++kc) {
        v8i a, bf;
#pragma unroll
        for (int v = 0; v < 8; ++v) {
            const int baseKA = ((v >> 1) * 16) + ((v & 1) * 4) + half * 8;
            a[v] = *(const int*)(scoreB + m * NSITES + kc * 64 + baseKA);

            const int baseKB = ((v >> 1) * 16) + ((v & 1) * 4) + half * 8;
            const int k0 = kc * 64 + baseKB;
            const uint32_t packed =
                ((uint32_t)counts[k0 + 0] & 0xFFu) |
                (((uint32_t)counts[k0 + 1] & 0xFFu) << 8) |
                (((uint32_t)counts[k0 + 2] & 0xFFu) << 16) |
                (((uint32_t)counts[k0 + 3] & 0xFFu) << 24);
            bf[v] = ((lane & 15) == 0) ? (int)packed : 0;
        }
        acc = __builtin_amdgcn_wmma_i32_16x16x64_iu8(false, a, false, bf, acc,
                                                     false, false);
    }

    if ((lane & 15) == 0) {
#pragma unroll
        for (int r = 0; r < 8; ++r)
            out[tile * 16 + half * 8 + r] = (float)acc[r];
    }
}

extern "C" void kernel_launch(void* const* d_in, const int* in_sizes, int n_in,
                              void* d_out, int out_size, void* d_ws, size_t ws_size,
                              hipStream_t stream) {
    const int* L           = (const int*)d_in[0];
    const int* site_counts = (const int*)d_in[1];
    const int* pairs       = (const int*)d_in[2];

    uint32_t* packedL = (uint32_t*)d_ws;
    uint8_t*  scoreB  = (uint8_t*)d_ws + TIPROWS * NWORDS * 4;

    pack_tips<<<(TIPROWS * NWORDS + 255) / 256, 256, 0, stream>>>(L, packedL);

    fitch_steps<<<BS, 32, NNODES * ROWU32 * 4, stream>>>(packedL, pairs, scoreB);

    wmma_reduce<<<1, 64, 0, stream>>>(scoreB, site_counts, (float*)d_out);
}
